// LSTMModel_21844203667640
// MI455X (gfx1250) — matrix operand order repaired and checked by reading, not compiled, not run on hardware
//
#include <hip/hip_runtime.h>

// LSTM fused kernel for MI455X (gfx1250, wave32, WMMA bf16 16x16x32).
// B=4096, T=64, F=512, H=64 (4H=256 gates), O=2.
// - Weights register-resident as bf16 WMMA B-fragments (persistent-RNN style)
// - x stream software-pipelined: global loads for t+1 issued before WMMA(t)
// - x loads non-temporal (single-pass 512MB stream; keep weights in L2)

#define B_SIZE   4096
#define T_STEPS  64
#define F_IN     512
#define H_DIM    64
#define G_DIM    256
#define O_DIM    2

#define M_TILE   16      // batch rows per workgroup
#define BLOCK    256     // 8 wave32s

// padded LDS strides (avoid 64-bank conflicts on stride-1KB fragment reads)
#define SX_STRIDE_B   1040   // 520 bf16 per row (512 + 8 pad), 16B aligned
#define SG_STRIDE_F   260    // 260 f32 per row (256 + 4 pad)
#define SH_STRIDE_H   72     // 72 bf16 per row (64 + 8 pad), 16B aligned

typedef __bf16 bf16_t;
typedef bf16_t       v16bf __attribute__((ext_vector_type(16)));
typedef float        v8f   __attribute__((ext_vector_type(8)));
typedef float        v4f   __attribute__((ext_vector_type(4)));   // native vec for NT loads
typedef unsigned int v8u   __attribute__((ext_vector_type(8)));

__device__ __forceinline__ unsigned int f2bf1(float x) {
  unsigned int u = __float_as_uint(x);
  return (u + 0x7FFFu + ((u >> 16) & 1u)) >> 16;   // RNE fp32 -> bf16
}
__device__ __forceinline__ unsigned int pack2bf(float lo, float hi) {
  return f2bf1(lo) | (f2bf1(hi) << 16);
}
__device__ __forceinline__ float bf2f(unsigned int us) {
  return __uint_as_float(us << 16);
}
__device__ __forceinline__ float fast_sigmoid(float x) {
  float e = __builtin_amdgcn_exp2f(-1.4426950408889634f * x);
  return __builtin_amdgcn_rcpf(1.0f + e);
}
__device__ __forceinline__ float fast_tanh(float x) {
  // tanh(x) = 1 - 2/(1 + e^{2x}); well-behaved at both tails
  float e = __builtin_amdgcn_exp2f(2.8853900817779268f * x);
  return 1.0f - 2.0f * __builtin_amdgcn_rcpf(1.0f + e);
}

__global__ __launch_bounds__(BLOCK, 1) void lstm_fused_kernel(
    const float* __restrict__ x,   const float* __restrict__ Wih,
    const float* __restrict__ Whh, const float* __restrict__ bih,
    const float* __restrict__ bhh, const float* __restrict__ Wfc,
    const float* __restrict__ bfc, float* __restrict__ out)
{
  __shared__ __align__(16) unsigned char  sX[M_TILE * SX_STRIDE_B]; // bf16 x tile
  __shared__ __align__(16) float          sG[M_TILE * SG_STRIDE_F]; // f32 gates
  __shared__ __align__(16) unsigned short sH[M_TILE * SH_STRIDE_H]; // bf16 h state

  const int tid  = threadIdx.x;
  const int lane = tid & 31;
  const int wv   = tid >> 5;       // wave id 0..7 -> owns gate cols [32*wv, 32*wv+32)
  const int half = lane >> 4;      // 0/1 (wave32 halves per WMMA layout)
  const int ln   = lane & 15;      // M index (A/C) or N index (B/C)
  const int b0   = blockIdx.x * M_TILE;

  // ---------------- persistent bf16 weight fragments in VGPRs ----------------
  // B fragments as packed here (KxN, 16-bit, contiguous k): lane n = L%16, half = L/16;
  // VGPR j halfwords hold K = 2j+16*half, 2j+1+16*half  => 16 consecutive K.
  v8u   wihF[2][16];   // 2 n-tiles x 16 k-steps (K=512)
  v8u   whhF[2][2];    // 2 n-tiles x 2 k-steps  (K=64)
  float bias[2];

  #pragma unroll
  for (int a = 0; a < 2; ++a) {
    const int g = wv * 32 + a * 16 + ln;               // gate column (0..255)
    bias[a] = bih[g] + bhh[g];
    #pragma unroll
    for (int kk = 0; kk < 16; ++kk) {
      const v4f* p4 = (const v4f*)(Wih + (size_t)g * F_IN + kk * 32 + half * 16);
      v4f w0 = p4[0], w1 = p4[1], w2 = p4[2], w3 = p4[3];
      v8u r;
      r[0] = pack2bf(w0.x, w0.y); r[1] = pack2bf(w0.z, w0.w);
      r[2] = pack2bf(w1.x, w1.y); r[3] = pack2bf(w1.z, w1.w);
      r[4] = pack2bf(w2.x, w2.y); r[5] = pack2bf(w2.z, w2.w);
      r[6] = pack2bf(w3.x, w3.y); r[7] = pack2bf(w3.z, w3.w);
      wihF[a][kk] = r;
    }
    #pragma unroll
    for (int kk = 0; kk < 2; ++kk) {
      const v4f* p4 = (const v4f*)(Whh + (size_t)g * H_DIM + kk * 32 + half * 16);
      v4f w0 = p4[0], w1 = p4[1], w2 = p4[2], w3 = p4[3];
      v8u r;
      r[0] = pack2bf(w0.x, w0.y); r[1] = pack2bf(w0.z, w0.w);
      r[2] = pack2bf(w1.x, w1.y); r[3] = pack2bf(w1.z, w1.w);
      r[4] = pack2bf(w2.x, w2.y); r[5] = pack2bf(w2.z, w2.w);
      r[6] = pack2bf(w3.x, w3.y); r[7] = pack2bf(w3.z, w3.w);
      whhF[a][kk] = r;
    }
  }

  // ---------------- init state: h = 0 (LDS bf16), c = 0 (regs) ----------------
  float cReg[4];
  #pragma unroll
  for (int r = 0; r < 4; ++r) cReg[r] = 0.0f;
  for (int i = tid; i < M_TILE * SH_STRIDE_H; i += BLOCK) sH[i] = 0;

  // ---------------- prologue: in-flight x registers for t=0 ----------------
  // thread's 8 fixed staging slots: idx = tid + 256*i -> (m = idx>>7, c4 = idx&127)
  v4f xr[8];
  #pragma unroll
  for (int i = 0; i < 8; ++i) {
    int idx = tid + BLOCK * i;
    int m = idx >> 7, c4 = idx & 127;
    const v4f* p = ((const v4f*)(x + ((size_t)(b0 + m) * T_STEPS) * F_IN)) + c4;
    xr[i] = __builtin_nontemporal_load(p);
  }
  __syncthreads();

  // ---------------- time loop ----------------
  #pragma unroll 1
  for (int t = 0; t < T_STEPS; ++t) {
    // pack the in-flight registers (loaded for this t) -> bf16 LDS
    #pragma unroll
    for (int i = 0; i < 8; ++i) {
      int idx = tid + BLOCK * i;
      int m = idx >> 7, c4 = idx & 127;
      uint2 pk; pk.x = pack2bf(xr[i].x, xr[i].y); pk.y = pack2bf(xr[i].z, xr[i].w);
      *((uint2*)sX + (m * (SX_STRIDE_B / 8) + c4)) = pk;
    }
    __syncthreads();   // barrier A: sX staged, sH(t-1) visible

    // issue next timestep's global loads now; waited on only at next pack
    {
      const int tn = (t + 1 < T_STEPS) ? (t + 1) : t;
      #pragma unroll
      for (int i = 0; i < 8; ++i) {
        int idx = tid + BLOCK * i;
        int m = idx >> 7, c4 = idx & 127;
        const v4f* p =
            ((const v4f*)(x + ((size_t)(b0 + m) * T_STEPS + tn) * F_IN)) + c4;
        xr[i] = __builtin_nontemporal_load(p);
      }
    }

    // gates accumulators seeded with (b_ih + b_hh): C layout n = L%16 for both halves
    v8f acc[2];
    #pragma unroll
    for (int a = 0; a < 2; ++a)
      #pragma unroll
      for (int vi = 0; vi < 8; ++vi) acc[a][vi] = bias[a];

    // gates += x_t @ W_ih^T   (16 k-steps of 32)
    // A layout (MxK, 16-bit): lane m = L%16; two contiguous 16B chunks per k-step
    // at byte offsets 64*kk + 32*half and +16.
    #pragma unroll
    for (int kk = 0; kk < 16; ++kk) {
      const uint4* pa = (const uint4*)sX + (ln * (SX_STRIDE_B / 16) + kk * 4 + 2 * half);
      uint4 q0 = pa[0], q1 = pa[1];
      v8u ar; ar[0]=q0.x; ar[1]=q0.y; ar[2]=q0.z; ar[3]=q0.w;
              ar[4]=q1.x; ar[5]=q1.y; ar[6]=q1.z; ar[7]=q1.w;
      v16bf av = __builtin_bit_cast(v16bf, ar);
      #pragma unroll
      for (int a = 0; a < 2; ++a)
        acc[a] = __builtin_amdgcn_wmma_f32_16x16x32_bf16(
            false, av, false, __builtin_bit_cast(v16bf, wihF[a][kk]),
            (short)0, acc[a], false, false);
    }

    // gates += h @ W_hh^T   (2 k-steps of 32)
    #pragma unroll
    for (int kk = 0; kk < 2; ++kk) {
      const uint4* pa = (const uint4*)sH + (ln * (SH_STRIDE_H * 2 / 16) + kk * 4 + 2 * half);
      uint4 q0 = pa[0], q1 = pa[1];
      v8u ar; ar[0]=q0.x; ar[1]=q0.y; ar[2]=q0.z; ar[3]=q0.w;
              ar[4]=q1.x; ar[5]=q1.y; ar[6]=q1.z; ar[7]=q1.w;
      v16bf av = __builtin_bit_cast(v16bf, ar);
      #pragma unroll
      for (int a = 0; a < 2; ++a)
        acc[a] = __builtin_amdgcn_wmma_f32_16x16x32_bf16(
            false, av, false, __builtin_bit_cast(v16bf, whhF[a][kk]),
            (short)0, acc[a], false, false);
    }

    // spill gates to LDS f32: C/D layout -> row m = v + 8*half, col n
    #pragma unroll
    for (int a = 0; a < 2; ++a) {
      const int n = wv * 32 + a * 16 + ln;
      #pragma unroll
      for (int vi = 0; vi < 8; ++vi)
        sG[(vi + 8 * half) * SG_STRIDE_F + n] = acc[a][vi];
    }
    __syncthreads();   // barrier B: gates complete, WMMA reads of sX/sH done

    // elementwise LSTM cell: i,f,g,o in PyTorch order along the 256 gate cols
    #pragma unroll
    for (int r = 0; r < 4; ++r) {
      int idx = tid + BLOCK * r;          // 0..1023 over 16 rows x 64 hidden
      int m = idx >> 6, j = idx & 63;
      const float* gr = sG + m * SG_STRIDE_F;
      float gi = fast_sigmoid(gr[j]);
      float gf = fast_sigmoid(gr[64  + j]);
      float gg = fast_tanh   (gr[128 + j]);
      float go = fast_sigmoid(gr[192 + j]);
      float c  = gf * cReg[r] + gi * gg;
      cReg[r]  = c;
      float h  = go * fast_tanh(c);
      sH[m * SH_STRIDE_H + j] = (unsigned short)f2bf1(h);
    }
  }
  __syncthreads();   // final h complete

  // ---------------- FC head: out = h_T @ W_fc^T + b_fc  (16x2 per WG) ----------------
  if (tid < M_TILE * O_DIM) {
    int m = tid >> 1, o = tid & 1;
    float s = bfc[o];
    #pragma unroll
    for (int j = 0; j < H_DIM; ++j)
      s += bf2f(sH[m * SH_STRIDE_H + j]) * Wfc[o * H_DIM + j];
    out[(size_t)(b0 + m) * O_DIM + o] = s;
  }
}

extern "C" void kernel_launch(void* const* d_in, const int* in_sizes, int n_in,
                              void* d_out, int out_size, void* d_ws, size_t ws_size,
                              hipStream_t stream) {
  (void)in_sizes; (void)n_in; (void)out_size; (void)d_ws; (void)ws_size;
  const float* x   = (const float*)d_in[0];
  const float* Wih = (const float*)d_in[1];
  const float* Whh = (const float*)d_in[2];
  const float* bih = (const float*)d_in[3];
  const float* bhh = (const float*)d_in[4];
  const float* Wfc = (const float*)d_in[5];
  const float* bfc = (const float*)d_in[6];
  float* out = (float*)d_out;

  dim3 grid(B_SIZE / M_TILE);   // 256 workgroups
  dim3 block(BLOCK);            // 8 wave32s
  hipLaunchKernelGGL(lstm_fused_kernel, grid, block, 0, stream,
                     x, Wih, Whh, bih, bhh, Wfc, bfc, out);
}
